// AdditiveAttentionDecoder_37830071943235
// MI455X (gfx1250) — matrix operand order repaired and checked by reading, not compiled, not run on hardware
//
#include <hip/hip_runtime.h>
#include <hip/hip_bf16.h>
#include <hip/hip_fp16.h>

// ---------------------------------------------------------------------------
// AdditiveAttentionDecoder for MI455X (gfx1250, wave32, WMMA).
// bf16 weights (fits 192MB L2) + v_wmma_f32_16x16x32_bf16 everywhere.
// GEMM register-blocked: MB x NB tiles per wave -> 8 WMMAs per 12 b128 loads
// for the big GEMMs (vs 1 WMMA per 4 loads unblocked).
// ---------------------------------------------------------------------------

typedef __bf16 bf16;
typedef __attribute__((ext_vector_type(16))) __bf16 bf16x16;
typedef __attribute__((ext_vector_type(8)))  __bf16 bf16x8;
typedef __attribute__((ext_vector_type(8)))  float  f32x8;

constexpr int Vv = 32000, Ee = 512, Hh = 1024, Aa = 1024, Ll = 4;
constexpr int Bb = 32, Tt = 64, Ss = 128;
constexpr int G3H = 3 * Hh;          // 3072 gate width
constexpr int KIN0 = Hh + Ee;        // 1536 layer-0 input width

static __device__ __forceinline__ float sigmoidf_(float x) {
  return 1.0f / (1.0f + __expf(-x));
}

// ---- WMMA fragment loaders (bf16, 16x16x32) -------------------------------
// A tile: rows = lane&15; lane<16 holds K {0..7,16..23}, lane>=16 holds
// K {8..15,24..31} (per CDNA5 ISA 16-bit A layout).
static __device__ __forceinline__ bf16x16 load_a_frag(const bf16* base, int ld) {
  int lane = threadIdx.x & 31;
  int half = lane >> 4;
  const bf16* p = base + (size_t)(lane & 15) * ld;
  bf16x8 lo = *(const bf16x8*)(p + half * 8);
  bf16x8 hi = *(const bf16x8*)(p + 16 + half * 8);
  bf16x16 a;
#pragma unroll
  for (int e = 0; e < 8; ++e) { a[e] = lo[e]; a[e + 8] = hi[e]; }
  return a;
}
// B tile from Bt (N x K row-major): column = lane&15 (a row of Bt),
// lane<16 holds K {0..7,16..23}, lane>=16 holds K {8..15,24..31} (matches A).
static __device__ __forceinline__ bf16x16 load_b_frag(const bf16* base, int ld) {
  int lane = threadIdx.x & 31;
  int half = lane >> 4;
  const bf16* p = base + (size_t)(lane & 15) * ld;
  bf16x8 lo = *(const bf16x8*)(p + half * 8);
  bf16x8 hi = *(const bf16x8*)(p + 16 + half * 8);
  bf16x16 b;
#pragma unroll
  for (int e = 0; e < 8; ++e) { b[e] = lo[e]; b[e + 8] = hi[e]; }
  return b;
}

// ---- register-blocked WMMA GEMM -------------------------------------------
// C(MxN, f32) = A(MxK, bf16) @ Bt(NxK, bf16)^T + bias.  Each wave computes an
// MB x NB block of 16x16 tiles.  remapB!=0: input row r = t*remapB + b stores
// to output row b*remapT + t (scan order -> (B,T,V)).
template <int MB, int NB>
__global__ __launch_bounds__(256) void k_gemm_bf16(
    const bf16* __restrict__ A, const bf16* __restrict__ Bt,
    const float* __restrict__ bias, float* __restrict__ C,
    int M, int N, int K, int remapB, int remapT) {
  int wave = (int)((blockIdx.x * blockDim.x + threadIdx.x) >> 5);
  int lane = threadIdx.x & 31;
  int gM = (M >> 4) / MB, gN = (N >> 4) / NB;
  if (wave >= gM * gN) return;             // wave-uniform: EXEC stays all-ones
  int wm = wave / gN, wn = wave % gN;
  const bf16* Ab = A + (size_t)wm * MB * 16 * K;
  const bf16* Bb = Bt + (size_t)wn * NB * 16 * K;

  const f32x8 zero = {0.f, 0.f, 0.f, 0.f, 0.f, 0.f, 0.f, 0.f};
  f32x8 acc[MB][NB];
#pragma unroll
  for (int i = 0; i < MB; ++i)
#pragma unroll
    for (int j = 0; j < NB; ++j) acc[i][j] = zero;

  for (int k = 0; k < K; k += 32) {
    bf16x16 af[MB];
#pragma unroll
    for (int i = 0; i < MB; ++i)
      af[i] = load_a_frag(Ab + (size_t)i * 16 * K + k, K);
#pragma unroll
    for (int j = 0; j < NB; ++j) {
      bf16x16 bf = load_b_frag(Bb + (size_t)j * 16 * K + k, K);
#pragma unroll
      for (int i = 0; i < MB; ++i)
        acc[i][j] = __builtin_amdgcn_wmma_f32_16x16x32_bf16(
            false, af[i], false, bf, (short)0, acc[i][j], false, false);
    }
  }

  int half = lane >> 4;
#pragma unroll
  for (int i = 0; i < MB; ++i) {
    int rbase = (wm * MB + i) * 16 + (half ? 8 : 0);
#pragma unroll
    for (int j = 0; j < NB; ++j) {
      int col = (wn * NB + j) * 16 + (lane & 15);
      float bv = bias ? bias[col] : 0.0f;
#pragma unroll
      for (int e = 0; e < 8; ++e) {
        int r = rbase + e;
        size_t off;
        if (remapB) {
          int tt = r / remapB, b = r % remapB;
          off = ((size_t)b * remapT + tt) * (size_t)N + col;
        } else {
          off = (size_t)r * N + col;
        }
        C[off] = acc[i][j][e] + bv;
      }
    }
  }
}

// ---- fp32 -> bf16 elementwise convert -------------------------------------
__global__ void k_cvt(const float* __restrict__ src, bf16* __restrict__ dst, size_t n) {
  size_t i = (size_t)blockIdx.x * blockDim.x + threadIdx.x;
  if (i < n) dst[i] = (bf16)src[i];
}

// ---- fp32 (KxN) -> bf16 transposed (NxK) ----------------------------------
__global__ void k_tcvt(const float* __restrict__ src, bf16* __restrict__ dst,
                       int K, int N) {
  size_t i = (size_t)blockIdx.x * blockDim.x + threadIdx.x;
  size_t tot = (size_t)K * N;
  if (i >= tot) return;
  int n = (int)(i / K), k = (int)(i % K);
  dst[i] = (bf16)src[(size_t)k * N + n];
}

// ---- embedding gather (B,T) -> bf16 (B,T,E) -------------------------------
__global__ void k_gather_emb(const int* __restrict__ x, const float* __restrict__ emb,
                             bf16* __restrict__ xe) {
  size_t i = (size_t)blockIdx.x * blockDim.x + threadIdx.x;
  size_t tot = (size_t)Bb * Tt * Ee;
  if (i >= tot) return;
  int bt = (int)(i / Ee), e = (int)(i % Ee);
  int tok = x[bt];
  xe[i] = (bf16)emb[(size_t)tok * Ee + e];  // row 0 is already zero (padding)
}

// ---- additive attention: scores = v . tanh(qWq + kproj), mask, softmax ----
// grid = B; 8 waves/block, one wave per score (s), lanes reduce over A=1024.
__global__ void k_scores_softmax(const float* __restrict__ qA,
                                 const float* __restrict__ kproj,
                                 const float* __restrict__ v,
                                 const unsigned char* __restrict__ mask,
                                 float* __restrict__ w) {
  int b = blockIdx.x;
  int lane = threadIdx.x & 31;
  int wv = threadIdx.x >> 5;
  __shared__ float sc[Ss];
  __shared__ float red[256];
  const float* q = qA + (size_t)b * Aa;
  const float* kp = kproj + (size_t)b * Ss * Aa;
  for (int s = wv; s < Ss; s += 8) {
    const float* kps = kp + (size_t)s * Aa;
    float acc = 0.f;
    for (int a = lane; a < Aa; a += 32)
      acc += v[a] * tanhf(q[a] + kps[a]);
#pragma unroll
    for (int off = 16; off > 0; off >>= 1)
      acc += __shfl_down(acc, off, 32);
    if (lane == 0) sc[s] = mask[(size_t)b * Ss + s] ? acc : -1e9f;
  }
  __syncthreads();
  float xv = (threadIdx.x < Ss) ? sc[threadIdx.x] : -1e30f;
  red[threadIdx.x] = xv;
  __syncthreads();
  for (int off = 128; off > 0; off >>= 1) {
    if ((int)threadIdx.x < off)
      red[threadIdx.x] = fmaxf(red[threadIdx.x], red[threadIdx.x + off]);
    __syncthreads();
  }
  float mx = red[0];
  __syncthreads();
  float ex = (threadIdx.x < Ss) ? __expf(xv - mx) : 0.f;
  red[threadIdx.x] = ex;
  __syncthreads();
  for (int off = 128; off > 0; off >>= 1) {
    if ((int)threadIdx.x < off) red[threadIdx.x] += red[threadIdx.x + off];
    __syncthreads();
  }
  float inv = 1.0f / red[0];
  if (threadIdx.x < Ss) w[(size_t)b * Ss + threadIdx.x] = ex * inv;
}

// ---- attn = sum_s w*enc; pack inp = [attn(bf16), x_t(bf16)] ---------------
__global__ void k_attn_pack(const float* __restrict__ w, const float* __restrict__ enc,
                            const bf16* __restrict__ xe, bf16* __restrict__ inp, int t) {
  int b = blockIdx.x;
  __shared__ float ws[Ss];
  if (threadIdx.x < Ss) ws[threadIdx.x] = w[(size_t)b * Ss + threadIdx.x];
  __syncthreads();
  const float* encb = enc + (size_t)b * Ss * Hh;
  for (int h = threadIdx.x; h < Hh; h += blockDim.x) {
    float acc = 0.f;
    for (int s = 0; s < Ss; ++s) acc += ws[s] * encb[(size_t)s * Hh + h];
    inp[(size_t)b * KIN0 + h] = (bf16)acc;
  }
  const bf16* xt = xe + ((size_t)b * Tt + t) * Ee;
  for (int e = threadIdx.x; e < Ee; e += blockDim.x)
    inp[(size_t)b * KIN0 + Hh + e] = xt[e];
}

// ---- GRU elementwise: r,z,n gates -> h_new (f32 state + bf16 copy) --------
__global__ void k_gru_elem(const float* __restrict__ g, const float* __restrict__ gh,
                           float* __restrict__ h, bf16* __restrict__ hbf,
                           bf16* __restrict__ outs /*nullable*/, int n) {
  int i = blockIdx.x * blockDim.x + threadIdx.x;
  if (i >= n) return;
  int b = i / Hh, hh = i % Hh;
  size_t gb = (size_t)b * G3H;
  float r = sigmoidf_(g[gb + hh] + gh[gb + hh]);
  float z = sigmoidf_(g[gb + Hh + hh] + gh[gb + Hh + hh]);
  float nn = tanhf(g[gb + 2 * Hh + hh] + r * gh[gb + 2 * Hh + hh]);
  float hn = (1.f - z) * nn + z * h[i];
  h[i] = hn;
  bf16 hb = (bf16)hn;
  hbf[i] = hb;
  if (outs) outs[i] = hb;
}

// ---------------------------------------------------------------------------
extern "C" void kernel_launch(void* const* d_in, const int* in_sizes, int n_in,
                              void* d_out, int out_size, void* d_ws, size_t ws_size,
                              hipStream_t stream) {
  (void)in_sizes; (void)n_in; (void)out_size; (void)ws_size;
  const int*           x    = (const int*)d_in[0];
  const unsigned char* mask = (const unsigned char*)d_in[1];
  const float* enc  = (const float*)d_in[2];
  const float* h0   = (const float*)d_in[3];
  const float* emb  = (const float*)d_in[4];
  const float* Wq   = (const float*)d_in[5];
  const float* Wk   = (const float*)d_in[6];
  const float* vat  = (const float*)d_in[7];
  const float* Wx0  = (const float*)d_in[8];
  const float* Wxr  = (const float*)d_in[9];
  const float* Wh   = (const float*)d_in[10];
  const float* bx   = (const float*)d_in[11];
  const float* bh   = (const float*)d_in[12];
  const float* Wout = (const float*)d_in[13];
  const float* bout = (const float*)d_in[14];

  float* y = (float*)d_out;
  float* hfin = y + (size_t)Bb * Tt * Vv;

  // workspace partition
  char* p = (char*)d_ws;
  auto alloc = [&](size_t bytes) -> char* {
    char* r = p; p += (bytes + 255) & ~(size_t)255; return r;
  };
  bf16* enc_bf  = (bf16*)alloc((size_t)Bb * Ss * Hh * 2);
  bf16* xe_bf   = (bf16*)alloc((size_t)Bb * Tt * Ee * 2);
  bf16* Wk_t    = (bf16*)alloc((size_t)Aa * Hh * 2);
  bf16* Wq_t    = (bf16*)alloc((size_t)Aa * Hh * 2);
  bf16* Wx0_t   = (bf16*)alloc((size_t)G3H * KIN0 * 2);
  bf16* Wxr_t   = (bf16*)alloc((size_t)(Ll - 1) * G3H * Hh * 2);
  bf16* Wh_t    = (bf16*)alloc((size_t)Ll * G3H * Hh * 2);
  bf16* Wout_t  = (bf16*)alloc((size_t)Vv * Hh * 2);
  float* kproj  = (float*)alloc((size_t)Bb * Ss * Aa * 4);
  float* qA     = (float*)alloc((size_t)Bb * Aa * 4);
  float* watt   = (float*)alloc((size_t)Bb * Ss * 4);
  bf16* inp_bf  = (bf16*)alloc((size_t)Bb * KIN0 * 2);
  float* gbuf   = (float*)alloc((size_t)Bb * G3H * 4);
  float* ghbuf  = (float*)alloc((size_t)Bb * G3H * 4);
  float* h_f    = (float*)alloc((size_t)Ll * Bb * Hh * 4);
  bf16* h_bf    = (bf16*)alloc((size_t)Ll * Bb * Hh * 2);
  bf16* outs_bf = (bf16*)alloc((size_t)Tt * Bb * Hh * 2);

  auto cvt = [&](const float* s, bf16* d, size_t n) {
    k_cvt<<<(unsigned)((n + 255) / 256), 256, 0, stream>>>(s, d, n);
  };
  auto tcvt = [&](const float* s, bf16* d, int K, int N) {
    size_t n = (size_t)K * N;
    k_tcvt<<<(unsigned)((n + 255) / 256), 256, 0, stream>>>(s, d, K, N);
  };
  // big GEMMs: 2x4 tile block / wave (8 WMMA per 12 b128 loads)
  auto gemm_big = [&](const bf16* A, const bf16* Bt, const float* bias, float* C,
                      int M, int N, int K, int rB, int rT) {
    int waves = ((M / 16) / 2) * ((N / 16) / 4);
    k_gemm_bf16<2, 4><<<(waves + 7) / 8, 256, 0, stream>>>(A, Bt, bias, C, M, N, K, rB, rT);
  };
  // latency-critical skinny GEMMs (M=32): 1x2 block keeps wave count high
  auto gemm_small = [&](const bf16* A, const bf16* Bt, const float* bias, float* C,
                        int M, int N, int K) {
    int waves = (M / 16) * ((N / 16) / 2);
    k_gemm_bf16<1, 2><<<(waves + 7) / 8, 256, 0, stream>>>(A, Bt, bias, C, M, N, K, 0, 0);
  };

  // ---- one-time prep (re-run each call; ~60M elements, trivial) -----------
  cvt(enc, enc_bf, (size_t)Bb * Ss * Hh);
  {
    size_t n = (size_t)Bb * Tt * Ee;
    k_gather_emb<<<(unsigned)((n + 255) / 256), 256, 0, stream>>>(x, emb, xe_bf);
  }
  tcvt(Wk, Wk_t, Hh, Aa);
  tcvt(Wq, Wq_t, Hh, Aa);
  tcvt(Wx0, Wx0_t, KIN0, G3H);
  for (int l = 0; l < Ll - 1; ++l)
    tcvt(Wxr + (size_t)l * Hh * G3H, Wxr_t + (size_t)l * G3H * Hh, Hh, G3H);
  for (int l = 0; l < Ll; ++l)
    tcvt(Wh + (size_t)l * Hh * G3H, Wh_t + (size_t)l * G3H * Hh, Hh, G3H);
  tcvt(Wout, Wout_t, Hh, Vv);

  hipMemcpyAsync(h_f, h0, (size_t)Ll * Bb * Hh * 4, hipMemcpyDeviceToDevice, stream);
  cvt(h0, h_bf, (size_t)Ll * Bb * Hh);

  // kproj = enc @ Wk : (B*S=4096, H=1024) x (H, A=1024)
  gemm_big(enc_bf, Wk_t, nullptr, kproj, Bb * Ss, Aa, Hh, 0, 0);

  // ---- sequential decode scan (T=64) --------------------------------------
  for (int t = 0; t < Tt; ++t) {
    // q @ Wq  (q = top-layer hidden, bf16)
    gemm_small(h_bf + (size_t)(Ll - 1) * Bb * Hh, Wq_t, nullptr, qA, Bb, Aa, Hh);
    k_scores_softmax<<<Bb, 256, 0, stream>>>(qA, kproj, vat, mask, watt);
    k_attn_pack<<<Bb, 256, 0, stream>>>(watt, enc, xe_bf, inp_bf, t);
    for (int l = 0; l < Ll; ++l) {
      const bf16* Ain = (l == 0) ? inp_bf : h_bf + (size_t)(l - 1) * Bb * Hh;
      int Kin = (l == 0) ? KIN0 : Hh;
      const bf16* Wxt = (l == 0) ? Wx0_t : Wxr_t + (size_t)(l - 1) * G3H * Hh;
      gemm_small(Ain, Wxt, bx + (size_t)l * G3H, gbuf, Bb, G3H, Kin);
      gemm_small(h_bf + (size_t)l * Bb * Hh, Wh_t + (size_t)l * G3H * Hh,
                 bh + (size_t)l * G3H, ghbuf, Bb, G3H, Hh);
      k_gru_elem<<<(Bb * Hh + 255) / 256, 256, 0, stream>>>(
          gbuf, ghbuf, h_f + (size_t)l * Bb * Hh, h_bf + (size_t)l * Bb * Hh,
          (l == Ll - 1) ? outs_bf + (size_t)t * Bb * Hh : nullptr, Bb * Hh);
    }
  }

  // ---- y = outs @ Wout + bout, stored as (B,T,V) --------------------------
  gemm_big(outs_bf, Wout_t, bout, y, Tt * Bb, Vv, Hh, /*remapB=*/Bb, /*remapT=*/Tt);

  // ---- h_final -------------------------------------------------------------
  hipMemcpyAsync(hfin, h_f, (size_t)Ll * Bb * Hh * 4, hipMemcpyDeviceToDevice, stream);
}
